// MambaLayer_77807627534531
// MI455X (gfx1250) — matrix operand order repaired and checked by reading, not compiled, not run on hardware
//
#include <hip/hip_runtime.h>
#include <math.h>

// ---------------------------------------------------------------- types
typedef __bf16 bf16;
typedef __attribute__((ext_vector_type(16))) __bf16 v16bf;
typedef __attribute__((ext_vector_type(8)))  __bf16 v8bf;
typedef __attribute__((ext_vector_type(8)))  float  v8f;

#define BSZ     2
#define LSEQ    4096
#define DMODEL  256
#define DINNER  512
#define DSTATE  16
#define DTRANK  16
#define DCONV   4
#define NTOK    (BSZ * LSEQ)        // 8192 tokens
#define NPAD_XP 64                  // x_proj output padded 48 -> 64
#define TCHUNK  64                  // timesteps staged in LDS per scan chunk

// round-to-nearest-even f32 -> bf16
__device__ __forceinline__ bf16 f2bf(float f) {
    unsigned u = __builtin_bit_cast(unsigned, f);
    unsigned r = u + 0x7FFFu + ((u >> 16) & 1u);
    unsigned short h = (unsigned short)(r >> 16);
    return __builtin_bit_cast(bf16, h);
}

// ---------------------------------------------------------------- weight converts
__global__ void f32_to_bf16_kernel(const float* __restrict__ src, bf16* __restrict__ dst, int n) {
    int i = blockIdx.x * blockDim.x + threadIdx.x;
    if (i < n) dst[i] = f2bf(src[i]);
}

// x_proj_w is (48, 512); pad rows 48..63 with zeros so GEMM N=64
__global__ void cvt_xproj_pad(const float* __restrict__ src, bf16* __restrict__ dst) {
    int i = blockIdx.x * blockDim.x + threadIdx.x;
    if (i >= NPAD_XP * DINNER) return;
    int row = i / DINNER;
    dst[i] = (row < (DTRANK + 2 * DSTATE)) ? f2bf(src[i]) : f2bf(0.0f);
}

// ---------------------------------------------------------------- layernorms
__global__ void layernorm_bf16_kernel(const float* __restrict__ x,
                                      const float* __restrict__ w,
                                      const float* __restrict__ b,
                                      bf16* __restrict__ out) {
    __shared__ float s1[DMODEL], s2[DMODEL];
    const int row = blockIdx.x, tid = threadIdx.x;
    float v = x[(size_t)row * DMODEL + tid];
    s1[tid] = v; s2[tid] = v * v;
    __syncthreads();
    for (int s = DMODEL / 2; s > 0; s >>= 1) {
        if (tid < s) { s1[tid] += s1[tid + s]; s2[tid] += s2[tid + s]; }
        __syncthreads();
    }
    float mu  = s1[0] * (1.0f / DMODEL);
    float var = s2[0] * (1.0f / DMODEL) - mu * mu;
    float inv = rsqrtf(var + 1e-5f);
    out[(size_t)row * DMODEL + tid] = f2bf((v - mu) * inv * w[tid] + b[tid]);
}

__global__ void residual_layernorm_bf16(const float* __restrict__ ym,
                                        const float* __restrict__ x,
                                        const float* __restrict__ skip,
                                        const float* __restrict__ w,
                                        const float* __restrict__ b,
                                        bf16* __restrict__ out) {
    __shared__ float s1[DMODEL], s2[DMODEL];
    const int row = blockIdx.x, tid = threadIdx.x;
    size_t idx = (size_t)row * DMODEL + tid;
    float v = ym[idx] + skip[0] * x[idx];
    s1[tid] = v; s2[tid] = v * v;
    __syncthreads();
    for (int s = DMODEL / 2; s > 0; s >>= 1) {
        if (tid < s) { s1[tid] += s1[tid + s]; s2[tid] += s2[tid + s]; }
        __syncthreads();
    }
    float mu  = s1[0] * (1.0f / DMODEL);
    float var = s2[0] * (1.0f / DMODEL) - mu * mu;
    float inv = rsqrtf(var + 1e-5f);
    out[idx] = f2bf((v - mu) * inv * w[tid] + b[tid]);
}

// ---------------------------------------------------------------- bf16 WMMA GEMM
// C[M,N] = A[M,K] * W[N,K]^T (+bias[n]); A,W row-major with K contiguous.
// Each wave owns a 16x64 strip of C: the A fragment is loaded once per K-step
// and reused for 4 v_wmma_f32_16x16x32_bf16 (4 N-tiles).
// ISA 7.12.2 lane layout:
//   A (16x32): lanes 0-15 -> M=lane, K {0..7,16..23}; lanes 16-31 -> M=lane-16, K {8..15,24..31}
//   B (32x16): N=lane&15; A and B are both loaded here with contiguous K: lanes 0-15 K 0..15, lanes 16-31 K 16..31
//   C/D: vgpr r: M = r + (lane<16 ? 0 : 8), N = lane&15
template <int K>
__global__ __launch_bounds__(256) void gemm_bf16_wmma(const bf16* __restrict__ A,
                                                      const bf16* __restrict__ W,
                                                      const float* __restrict__ bias,
                                                      float* __restrict__ C,
                                                      int M, int N) {
    const int lane = threadIdx.x & 31;
    const int wave = blockIdx.x * (blockDim.x >> 5) + (threadIdx.x >> 5);
    const int groupsN = N >> 6;               // groups of 4 N-tiles
    const int tilesM  = M >> 4;
    if (wave >= tilesM * groupsN) return;     // wave-uniform; EXEC stays all-ones
    const int tm = wave / groupsN;
    const int tg = wave - tm * groupsN;

    const int half = lane >> 4;               // 0: lanes 0-15, 1: lanes 16-31
    const int l16  = lane & 15;
    const bf16* Ap = A + (size_t)(tm * 16 + l16) * K + half * 16;
    const bf16* Wp = W + (size_t)(tg * 64 + l16) * K + half * 16;

    v8f acc[4] = {{}, {}, {}, {}};
#pragma unroll
    for (int kb = 0; kb < K; kb += 32) {
        v8bf a0 = *(const v8bf*)(Ap + kb);        // K 0..7  (or 16..23)
        v8bf a1 = *(const v8bf*)(Ap + kb + 8);    // K 8..15 (or 24..31)
        v16bf a;
#pragma unroll
        for (int i = 0; i < 8; ++i) { a[i] = a0[i]; a[i + 8] = a1[i]; }
#pragma unroll
        for (int t = 0; t < 4; ++t) {
            v16bf bm = *(const v16bf*)(Wp + (size_t)t * 16 * K + kb);
            acc[t] = __builtin_amdgcn_wmma_f32_16x16x32_bf16(false, a, false, bm,
                                                             (short)0, acc[t], false, false);
        }
    }

    const int mbase = tm * 16 + half * 8;
#pragma unroll
    for (int t = 0; t < 4; ++t) {
        const int n = tg * 64 + t * 16 + l16;
        float bv = bias ? bias[n] : 0.0f;
#pragma unroll
        for (int r = 0; r < 8; ++r)
            C[(size_t)(mbase + r) * N + n] = acc[t][r] + bv;
    }
}

// ---------------------------------------------------------------- depthwise conv + silu
__global__ void conv_silu(const float* __restrict__ xz,    // (NTOK, 1024), xi = cols 0..511
                          const float* __restrict__ cw,    // (512,1,4)
                          const float* __restrict__ cb,
                          float* __restrict__ xc,          // (NTOK, 512) f32
                          bf16* __restrict__ xcb) {        // (NTOK, 512) bf16
    int i = blockIdx.x * blockDim.x + threadIdx.x;
    if (i >= NTOK * DINNER) return;
    int d  = i & (DINNER - 1);
    int bt = i >> 9;
    int t  = bt & (LSEQ - 1);
    int b  = bt >> 12;
    float acc = cb[d];
#pragma unroll
    for (int j = 0; j < DCONV; ++j) {
        int tj = t + j - (DCONV - 1);
        if (tj >= 0)
            acc += cw[d * DCONV + j] * xz[(size_t)(b * LSEQ + tj) * (2 * DINNER) + d];
    }
    float s = acc / (1.0f + __expf(-acc));   // silu
    xc[i]  = s;
    xcb[i] = f2bf(s);
}

// ---------------------------------------------------------------- delta = softplus(dt @ W^T + b)
__global__ void delta_softplus(const float* __restrict__ dbc,   // (NTOK, 64): dt|B|C|pad
                               const float* __restrict__ dtw,   // (512, 16)
                               const float* __restrict__ dtb,
                               float* __restrict__ delta) {     // (NTOK, 512)
    __shared__ float dts[DTRANK];
    const int row = blockIdx.x, d = threadIdx.x;
    if (d < DTRANK) dts[d] = dbc[(size_t)row * NPAD_XP + d];
    __syncthreads();
    float acc = dtb[d];
#pragma unroll
    for (int r = 0; r < DTRANK; ++r) acc += dts[r] * dtw[d * DTRANK + r];
    float sp = (acc > 20.0f) ? acc : log1pf(__expf(acc));
    delta[(size_t)row * DINNER + d] = sp;
}

// ---------------------------------------------------------------- selective scan
// One LANE per (b, d, n) state: the h[t] = a*h + x recurrence is independent
// per (d,n); y = sum_n h*C is a 4-step shfl_xor butterfly over the 16-lane
// n-group (off the serial h chain). 16 channels per block (1 per 16 lanes),
// 512 waves total across the grid. delta/u/B/C staged in LDS per 64-step chunk.
__global__ __launch_bounds__(256) void selective_scan(const float* __restrict__ delta,
                                                      const float* __restrict__ u,     // xc
                                                      const float* __restrict__ dbc,
                                                      const float* __restrict__ A_log,
                                                      const float* __restrict__ Dsk,
                                                      float* __restrict__ ys) {
    __shared__ float BC[TCHUNK * 32];        // per-t: [0..15]=B, [16..31]=C
    __shared__ float DT[TCHUNK * 16];        // per-t: delta for this block's 16 channels
    __shared__ float UU[TCHUNK * 16];        // per-t: u     for this block's 16 channels

    const int tid = threadIdx.x;
    const int n   = tid & 15;                // state index within channel
    const int ch  = tid >> 4;                // channel 0..15 within block
    const int dbase = blockIdx.x * 16;
    const int d   = dbase + ch;
    const int b   = blockIdx.y;

    const float A  = -__expf(A_log[d * DSTATE + n]);
    const float Dd = Dsk[d];
    float h = 0.0f;

    for (int t0 = 0; t0 < LSEQ; t0 += TCHUNK) {
        const size_t rowbase = (size_t)(b * LSEQ + t0);
        for (int e = tid; e < TCHUNK * 32; e += blockDim.x) {
            int tt = e >> 5, c = e & 31;
            BC[e] = dbc[(rowbase + tt) * NPAD_XP + DTRANK + c];
        }
        for (int e = tid; e < TCHUNK * 16; e += blockDim.x) {
            int tt = e >> 4, c = e & 15;
            size_t idx = (rowbase + tt) * DINNER + dbase + c;
            DT[e] = delta[idx];
            UU[e] = u[idx];
        }
        __syncthreads();
        for (int tt = 0; tt < TCHUNK; ++tt) {
            float dt = DT[tt * 16 + ch];
            float ut = UU[tt * 16 + ch];
            h = __expf(dt * A) * h + (dt * ut) * BC[tt * 32 + n];
            float p = h * BC[tt * 32 + 16 + n];
            p += __shfl_xor(p, 1);
            p += __shfl_xor(p, 2);
            p += __shfl_xor(p, 4);
            p += __shfl_xor(p, 8);
            if (n == 0)
                ys[(rowbase + tt) * DINNER + d] = p + ut * Dd;
        }
        __syncthreads();
    }
}

// ---------------------------------------------------------------- gate: y * silu(z) -> bf16
__global__ void gate_silu_bf16(const float* __restrict__ ys,
                               const float* __restrict__ xz,   // z = cols 512..1023
                               bf16* __restrict__ yg) {
    int i = blockIdx.x * blockDim.x + threadIdx.x;
    if (i >= NTOK * DINNER) return;
    int bt = i >> 9, d = i & (DINNER - 1);
    float z = xz[(size_t)bt * (2 * DINNER) + DINNER + d];
    float g = z / (1.0f + __expf(-z));
    yg[i] = f2bf(ys[i] * g);
}

// ---------------------------------------------------------------- host launch
extern "C" void kernel_launch(void* const* d_in, const int* in_sizes, int n_in,
                              void* d_out, int out_size, void* d_ws, size_t ws_size,
                              hipStream_t stream) {
    const float* x          = (const float*)d_in[0];
    const float* ln_w       = (const float*)d_in[1];
    const float* ln_b       = (const float*)d_in[2];
    const float* in_proj_w  = (const float*)d_in[3];
    const float* conv_w     = (const float*)d_in[4];
    const float* conv_b     = (const float*)d_in[5];
    const float* x_proj_w   = (const float*)d_in[6];
    const float* dt_proj_w  = (const float*)d_in[7];
    const float* dt_proj_b  = (const float*)d_in[8];
    const float* A_log      = (const float*)d_in[9];
    const float* Dskip      = (const float*)d_in[10];
    const float* out_proj_w = (const float*)d_in[11];
    const float* proj_w     = (const float*)d_in[12];
    const float* proj_b     = (const float*)d_in[13];
    const float* skip_scale = (const float*)d_in[14];
    float* out = (float*)d_out;

    char* ws = (char*)d_ws;
    size_t off = 0;
    auto alloc = [&](size_t bytes) -> void* {
        void* p = ws + off;
        off = (off + bytes + 255) & ~(size_t)255;
        return p;
    };

    bf16*  w_in  = (bf16*) alloc((size_t)(2 * DINNER) * DMODEL * 2);   // 1024x256
    bf16*  w_xp  = (bf16*) alloc((size_t)NPAD_XP * DINNER * 2);        // 64x512 (padded)
    bf16*  w_out = (bf16*) alloc((size_t)DMODEL * DINNER * 2);         // 256x512
    bf16*  w_pj  = (bf16*) alloc((size_t)DMODEL * DMODEL * 2);         // 256x256
    bf16*  xn    = (bf16*) alloc((size_t)NTOK * DMODEL * 2);           // LN(x), reused as xm
    float* xz    = (float*)alloc((size_t)NTOK * 2 * DINNER * 4);       // in_proj out
    float* xc    = (float*)alloc((size_t)NTOK * DINNER * 4);           // conv+silu f32
    bf16*  xcb   = (bf16*) alloc((size_t)NTOK * DINNER * 2);           // conv+silu bf16
    float* dbc   = (float*)alloc((size_t)NTOK * NPAD_XP * 4);          // x_proj out
    float* delta = (float*)alloc((size_t)NTOK * DINNER * 4);
    float* ysb   = (float*)alloc((size_t)NTOK * DINNER * 4);           // scan out
    bf16*  yg    = (bf16*) alloc((size_t)NTOK * DINNER * 2);           // gated bf16
    float* ym    = (float*)alloc((size_t)NTOK * DMODEL * 4);           // out_proj out

    // weight conversions
    f32_to_bf16_kernel<<<(2 * DINNER * DMODEL + 255) / 256, 256, 0, stream>>>(in_proj_w, w_in, 2 * DINNER * DMODEL);
    cvt_xproj_pad<<<(NPAD_XP * DINNER + 255) / 256, 256, 0, stream>>>(x_proj_w, w_xp);
    f32_to_bf16_kernel<<<(DMODEL * DINNER + 255) / 256, 256, 0, stream>>>(out_proj_w, w_out, DMODEL * DINNER);
    f32_to_bf16_kernel<<<(DMODEL * DMODEL + 255) / 256, 256, 0, stream>>>(proj_w, w_pj, DMODEL * DMODEL);

    // LN1 -> bf16
    layernorm_bf16_kernel<<<NTOK, DMODEL, 0, stream>>>(x, ln_w, ln_b, xn);

    // in_proj: (8192,1024) = xn(8192,256) x W(1024,256)^T
    {
        int waves = (NTOK / 16) * ((2 * DINNER) / 64);
        gemm_bf16_wmma<DMODEL><<<(waves + 7) / 8, 256, 0, stream>>>(xn, w_in, nullptr, xz, NTOK, 2 * DINNER);
    }

    // depthwise causal conv + bias + silu
    conv_silu<<<(NTOK * DINNER + 255) / 256, 256, 0, stream>>>(xz, conv_w, conv_b, xc, xcb);

    // x_proj: (8192,64) = xc(8192,512) x W(64,512)^T
    {
        int waves = (NTOK / 16) * (NPAD_XP / 64);
        gemm_bf16_wmma<DINNER><<<(waves + 7) / 8, 256, 0, stream>>>(xcb, w_xp, nullptr, dbc, NTOK, NPAD_XP);
    }

    // delta = softplus(dt @ dt_proj_w^T + dt_proj_b)
    delta_softplus<<<NTOK, DINNER, 0, stream>>>(dbc, dt_proj_w, dt_proj_b, delta);

    // selective scan over L: one lane per (b,d,n)
    {
        dim3 grid(DINNER / 16, BSZ);
        selective_scan<<<grid, 256, 0, stream>>>(delta, xc, dbc, A_log, Dskip, ysb);
    }

    // y * silu(z) -> bf16
    gate_silu_bf16<<<(NTOK * DINNER + 255) / 256, 256, 0, stream>>>(ysb, xz, yg);

    // out_proj: (8192,256) = yg(8192,512) x W(256,512)^T
    {
        int waves = (NTOK / 16) * (DMODEL / 64);
        gemm_bf16_wmma<DINNER><<<(waves + 7) / 8, 256, 0, stream>>>(yg, w_out, nullptr, ym, NTOK, DMODEL);
    }

    // residual + LN2 -> bf16 (reuse xn)
    residual_layernorm_bf16<<<NTOK, DMODEL, 0, stream>>>(ym, x, skip_scale, ln_w, ln_b, xn);

    // final proj with bias -> fp32 output
    {
        int waves = (NTOK / 16) * (DMODEL / 64);
        gemm_bf16_wmma<DMODEL><<<(waves + 7) / 8, 256, 0, stream>>>(xn, w_pj, proj_b, out, NTOK, DMODEL);
    }
}
